// GNN_14465449853443
// MI455X (gfx1250) — matrix operand order repaired and checked by reading, not compiled, not run on hardware
//
#include <hip/hip_runtime.h>
#include <hip/hip_bf16.h>

#define HDIM 128

typedef __attribute__((ext_vector_type(16))) __bf16 v16bf;
typedef __attribute__((ext_vector_type(8)))  float  v8f;

// ---------------------------------------------------------------------------
// Zero a float region (d_ws is poisoned 0xAA; must re-zero every call).
// ---------------------------------------------------------------------------
__global__ __launch_bounds__(256) void zero_f32(float* __restrict__ p, long long n) {
    long long i = (long long)blockIdx.x * 256 + threadIdx.x;
    if (i < n) p[i] = 0.0f;
}

// ---------------------------------------------------------------------------
// Pre-pack a 128x128 f32 weight matrix into bf16 WMMA B-fragment order.
// Packed index t = ((ct*4 + ks)*32 + lane)*16 + e  maps to
//   W[(ct*16 + lane%16)*128 + ks*32 + (e/8)*16 + (lane/16)*8 + e%8]
// so each GEMM wave reads its fragment as one aligned 32B vector load.
// ---------------------------------------------------------------------------
__global__ __launch_bounds__(256) void pack_w(const float* __restrict__ W,
                                              __bf16* __restrict__ P) {
    int t = blockIdx.x * 256 + threadIdx.x;        // 0 .. 16383
    int e    = t & 15;                             // K sub-offset within fragment
    int lane = (t >> 4) & 31;
    int ks   = (t >> 9) & 3;                       // K-step (k0/32)
    int ct   = (t >> 11) & 7;                      // column tile (wave)
    int n = ct * 16 + (lane & 15);
    int k = ks * 32 + ((e & 8) << 1) + (lane >> 4) * 8 + (e & 7);
    P[t] = (__bf16)W[n * HDIM + k];
}

// ---------------------------------------------------------------------------
// Per-destination in-degree counts (float, matches reference segment_sum of 1s)
// ---------------------------------------------------------------------------
__global__ __launch_bounds__(256) void count_deg(const long long* __restrict__ ei,
                                                 float* __restrict__ cnt,
                                                 long long nE) {
    long long e = (long long)blockIdx.x * 256 + threadIdx.x;
    if (e < nE) {
        int dst = (int)ei[nE + e];
        atomicAdd(&cnt[dst], 1.0f);
    }
}

// ---------------------------------------------------------------------------
// Edge scatter-add: one wave32 per edge; lane l moves 4 contiguous floats.
// 32 lanes x float4 = full 512B feature row: coalesced gather + L2 atomics
// (feature matrix is 51MB -> resident in the 192MB L2).
// ---------------------------------------------------------------------------
__global__ __launch_bounds__(256) void scatter_sum(const float* __restrict__ x,
                                                   const long long* __restrict__ ei,
                                                   float* __restrict__ sum,
                                                   long long nE) {
    long long gid = (long long)blockIdx.x * 256 + threadIdx.x;
    long long e   = gid >> 5;
    int lane      = (int)(gid & 31);
    if (e >= nE) return;
    int src = (int)ei[e];
    int dst = (int)ei[nE + e];
    const float4 v = *(const float4*)(x + (long long)src * HDIM + lane * 4);
    float* d = sum + (long long)dst * HDIM + lane * 4;
    atomicAdd(d + 0, v.x);
    atomicAdd(d + 1, v.y);
    atomicAdd(d + 2, v.z);
    atomicAdd(d + 3, v.w);
}

// ---------------------------------------------------------------------------
// Fused SAGE linear:  out = relu?( mean @ Wl^T + bl + xin @ Wr^T )
// mean computed on the fly from sum/cnt. bf16 WMMA 16x16x32, f32 accum.
// Block = 8 waves; block owns 16 rows, wave w owns cols [16w,16w+16).
// Weights arrive pre-packed in B-fragment order (Pl, Pr).
// ---------------------------------------------------------------------------
__global__ __launch_bounds__(256) void sage_gemm(const float* __restrict__ sum,
                                                 const float* __restrict__ cnt,
                                                 const float* __restrict__ xin,
                                                 const __bf16* __restrict__ Pl,
                                                 const float* __restrict__ bl,
                                                 const __bf16* __restrict__ Pr,
                                                 float* __restrict__ out,
                                                 int n, int relu) {
    __shared__ __bf16 sMean[16][HDIM];
    __shared__ __bf16 sX[16][HDIM];

    const int r0 = blockIdx.x * 16;
    const int t  = threadIdx.x;

    // Stage 16 rows of (mean, x) into LDS as bf16. 2048 elems / 256 threads.
#pragma unroll
    for (int i = 0; i < 8; ++i) {
        int e   = t + 256 * i;
        int row = e >> 7;            // /128
        int col = e & (HDIM - 1);
        int gr  = r0 + row;
        float m = 0.0f, xv = 0.0f;
        if (gr < n) {
            float c = cnt[gr];
            c  = (c > 1.0f) ? c : 1.0f;
            m  = sum[(long long)gr * HDIM + col] / c;
            xv = xin[(long long)gr * HDIM + col];
        }
        sMean[row][col] = (__bf16)m;
        sX[row][col]    = (__bf16)xv;
    }
    __syncthreads();

    const int wave = t >> 5;
    const int lane = t & 31;
    const int half = lane >> 4;                 // 0: lanes 0-15, 1: lanes 16-31
    const int mrow = lane & 15;                 // A-fragment row (M)
    const int gcol = wave * 16 + (lane & 15);   // B/D column (N)

    const v16bf* PL = (const v16bf*)Pl;
    const v16bf* PR = (const v16bf*)Pr;

    v8f acc = {};

#pragma unroll
    for (int ks = 0; ks < 4; ++ks) {            // K-steps of 32
        const int k0 = ks * 32;
        v16bf aM, aX;
        // A layout (16-bit, 16x32): VGPR v holds K = (v<4?0:16) + half*8 + (v%4)*2 + {0,1}
#pragma unroll
        for (int v = 0; v < 8; ++v) {
            int kk = k0 + ((v & 4) << 2) + half * 8 + (v & 3) * 2;
            aM[2 * v]     = sMean[mrow][kk];
            aM[2 * v + 1] = sMean[mrow][kk + 1];
            aX[2 * v]     = sX[mrow][kk];
            aX[2 * v + 1] = sX[mrow][kk + 1];
        }
        // Pre-packed bf16 B-fragments: one aligned 32B load per matrix.
        const v16bf bL = PL[(wave * 4 + ks) * 32 + lane];
        const v16bf bR = PR[(wave * 4 + ks) * 32 + lane];

        acc = __builtin_amdgcn_wmma_f32_16x16x32_bf16(false, aM, false, bL,
                                                      (short)0, acc, false, false);
        acc = __builtin_amdgcn_wmma_f32_16x16x32_bf16(false, aX, false, bR,
                                                      (short)0, acc, false, false);
    }

    // Epilogue: bias (lin_l only) + optional ReLU.
    // D layout: VGPR r -> row M = r + 8*half, col N = lane%16.
    const float bias = bl[gcol];
#pragma unroll
    for (int r = 0; r < 8; ++r) {
        int grow = r0 + r + half * 8;
        if (grow < n) {
            float v = acc[r] + bias;
            if (relu) v = (v > 0.0f) ? v : 0.0f;
            out[(long long)grow * HDIM + gcol] = v;
        }
    }
}

// ---------------------------------------------------------------------------
extern "C" void kernel_launch(void* const* d_in, const int* in_sizes, int n_in,
                              void* d_out, int out_size, void* d_ws, size_t ws_size,
                              hipStream_t stream) {
    const float*     x   = (const float*)d_in[0];
    const long long* ei  = (const long long*)d_in[1];
    const float*     W1l = (const float*)d_in[2];
    const float*     b1l = (const float*)d_in[3];
    const float*     W1r = (const float*)d_in[4];
    const float*     W2l = (const float*)d_in[5];
    const float*     b2l = (const float*)d_in[6];
    const float*     W2r = (const float*)d_in[7];
    float*           out = (float*)d_out;

    const long long N = in_sizes[0] / HDIM;
    const long long E = in_sizes[1] / 2;
    const long long WPACK = (long long)HDIM * HDIM;   // 16384 bf16 per matrix

    // Workspace layout: sum[N*H] f32 | cnt[N] f32 | 4x packed bf16 weights (32B aligned)
    float* sum = (float*)d_ws;
    float* cnt = sum + N * HDIM;
    uintptr_t wb = ((uintptr_t)(cnt + N) + 31) & ~(uintptr_t)31;
    __bf16* P1l = (__bf16*)wb;
    __bf16* P1r = P1l + WPACK;
    __bf16* P2l = P1r + WPACK;
    __bf16* P2r = P2l + WPACK;

    const unsigned zb_all = (unsigned)((N * HDIM + N + 255) / 256);
    const unsigned zb_sum = (unsigned)((N * HDIM + 255) / 256);
    const unsigned cb     = (unsigned)((E + 255) / 256);
    const unsigned sb     = (unsigned)((E * 32 + 255) / 256);
    const unsigned gb     = (unsigned)((N + 15) / 16);
    const unsigned pb     = (unsigned)(WPACK / 256);

    // --- Prep: pack all four weight matrices into WMMA B-fragment bf16 ---
    pack_w<<<pb, 256, 0, stream>>>(W1l, P1l);
    pack_w<<<pb, 256, 0, stream>>>(W1r, P1r);
    pack_w<<<pb, 256, 0, stream>>>(W2l, P2l);
    pack_w<<<pb, 256, 0, stream>>>(W2r, P2r);

    // --- Layer 1 ---
    zero_f32<<<zb_all, 256, 0, stream>>>(sum, N * HDIM + N);   // sum + cnt contiguous
    count_deg<<<cb, 256, 0, stream>>>(ei, cnt, E);
    scatter_sum<<<sb, 256, 0, stream>>>(x, ei, sum, E);
    // h1 lives in d_out (each block stages its rows to LDS before overwriting them)
    sage_gemm<<<gb, 256, 0, stream>>>(sum, cnt, x, P1l, b1l, P1r, out, (int)N, 1);

    // --- Layer 2 ---
    zero_f32<<<zb_sum, 256, 0, stream>>>(sum, N * HDIM);       // keep cnt
    scatter_sum<<<sb, 256, 0, stream>>>(out, ei, sum, E);
    sage_gemm<<<gb, 256, 0, stream>>>(sum, cnt, out, P2l, b2l, P2r, out, (int)N, 0);
}
